// kgLinear_11252814315778
// MI455X (gfx1250) — matrix operand order repaired and checked by reading, not compiled, not run on hardware
//
#include <hip/hip_runtime.h>
#include <hip/hip_bf16.h>

// ---- sizes from the reference ----
#define IN_SIZE   1024
#define OUT_SIZE  1024
#define KDIM      4
#define NGROUPS   32
#define VSEL      16
#define BATCH     256

// ---- blocking ----
#define KC        32                 // fp32 K-elements staged per chunk
#define OBLK      128                // output columns per workgroup
#define NCHUNK    (IN_SIZE / KC)     // 32
#define ROWB      144                // LDS row stride bytes: 128 data + 16 pad (TDM pad)
#define XROWS     BATCH              // 256
#define WROWS     OBLK               // 128
#define XBYTES    (XROWS * ROWB)     // 36864
#define BUFBYTES  (XBYTES + WROWS * ROWB) // 55296
#define SMEM_TOTAL (2 * BUFBYTES)    // 110592 (double buffered)

typedef __bf16 bf16;
typedef __attribute__((ext_vector_type(8)))  bf16         bf16x8;
typedef __attribute__((ext_vector_type(16))) bf16         v16bf;
typedef __attribute__((ext_vector_type(8)))  float        v8f;
typedef __attribute__((ext_vector_type(4)))  float        float4v;
typedef __attribute__((ext_vector_type(4)))  unsigned int uint4v;
typedef __attribute__((ext_vector_type(8)))  int          int8v;
typedef __attribute__((ext_vector_type(4)))  int          int4v;

#if defined(__gfx1250__)

#if __has_builtin(__builtin_amdgcn_tensor_load_to_lds)
#define HAS_TDM 1
#else
#define HAS_TDM 0
#endif

__device__ __forceinline__ unsigned pk_bf16(bf16 a, bf16 b) {
  union { bf16 h[2]; unsigned u; } u_;
  u_.h[0] = a; u_.h[1] = b;
  return u_.u;
}

// In-place: row of 32 fp32 (128B) -> [32 bf16 hi | 32 bf16 lo] (64B + 64B).
// All LDS traffic via uint4v to keep one alias type; same-thread RAW is safe.
__device__ __forceinline__ void cvt_row(char* row) {
  uint4v* r4 = (uint4v*)row;
  float f[32];
#pragma unroll
  for (int j = 0; j < 8; ++j) {
    uint4v u = r4[j];
    f[4*j+0] = __builtin_bit_cast(float, (unsigned)u[0]);
    f[4*j+1] = __builtin_bit_cast(float, (unsigned)u[1]);
    f[4*j+2] = __builtin_bit_cast(float, (unsigned)u[2]);
    f[4*j+3] = __builtin_bit_cast(float, (unsigned)u[3]);
  }
  unsigned hi[16], lo[16];
#pragma unroll
  for (int j = 0; j < 16; ++j) {
    float a = f[2*j], b = f[2*j+1];
    bf16 ha = (bf16)a, hb = (bf16)b;                 // RNE via v_cvt
    bf16 la = (bf16)(a - (float)ha);
    bf16 lb = (bf16)(b - (float)hb);
    hi[j] = pk_bf16(ha, hb);
    lo[j] = pk_bf16(la, lb);
  }
#pragma unroll
  for (int j = 0; j < 4; ++j) {
    uint4v h = { hi[4*j], hi[4*j+1], hi[4*j+2], hi[4*j+3] };
    uint4v l = { lo[4*j], lo[4*j+1], lo[4*j+2], lo[4*j+3] };
    r4[j]     = h;   // bytes [0,64)
    r4[4 + j] = l;   // bytes [64,128)
  }
}

__device__ __forceinline__ v16bf load_frag(const char* base, int off1, int off2) {
  bf16x8 a = *(const bf16x8*)(base + off1);
  bf16x8 b = *(const bf16x8*)(base + off2);
  return __builtin_shufflevector(a, b, 0,1,2,3,4,5,6,7,8,9,10,11,12,13,14,15);
}

#if HAS_TDM
// 2D TDM tile load: tile_d1 rows x tile_d0 fp32 elements, global row stride
// `stride0` (elements). Pads LDS dest 16B per 128B row -> ROWB=144 row stride.
__device__ __forceinline__ void tdm_load_2d(unsigned lds_off, const void* gaddr,
                                            unsigned tile_d0, unsigned tile_d1,
                                            unsigned long long stride0) {
  unsigned long long ga = (unsigned long long)gaddr;
  uint4v g0;
  g0[0] = 1u;                                   // count=1 (valid descriptor)
  g0[1] = lds_off;                              // lds_addr
  g0[2] = (unsigned)ga;                         // global_addr[31:0]
  g0[3] = (unsigned)((ga >> 32) & 0x01FFFFFFu)  // global_addr[56:32]
          | 0x80000000u;                        // type=2 ("image")
  int8v g1;
  // data_size=4B (2<<16) | pad_enable (1<<20) | pad_interval=32 dw (4<<22)
  // | pad_amount=4 dw (3<<25); workgroup_mask=0 (not in cluster)
  g1[0] = (int)(0x00020000u | 0x00100000u | (4u << 22) | (3u << 25));
  g1[1] = (int)((tile_d0 & 0xFFFFu) << 16);             // tensor_dim0[15:0]
  g1[2] = (int)((tile_d0 >> 16) | ((tile_d1 & 0xFFFFu) << 16)); // dim0 hi | dim1 lo
  g1[3] = (int)((tile_d1 >> 16) | (tile_d0 << 16));     // dim1 hi | tile_dim0
  g1[4] = (int)(tile_d1 & 0xFFFFu);                     // tile_dim1 | tile_dim2=0
  g1[5] = (int)(unsigned)(stride0 & 0xFFFFFFFFull);     // tensor_dim0_stride lo
  g1[6] = (int)(unsigned)((stride0 >> 32) & 0xFFFFull); // stride hi | dim1_stride=0
  g1[7] = 0;
  int4v gz4 = {0, 0, 0, 0};                             // 2D: groups 2/3 zero
  int8v gz8 = {0, 0, 0, 0, 0, 0, 0, 0};
  __builtin_amdgcn_tensor_load_to_lds(g0, g1, gz4, gz4, gz8, 0);
}
#endif // HAS_TDM
#endif // __gfx1250__

__global__ __launch_bounds__(256)
void kgrouped_linear_wmma(const float* __restrict__ x,
                          const float* __restrict__ w,
                          const float* __restrict__ bias,
                          const int*   __restrict__ gidx,
                          float*       __restrict__ out) {
#if defined(__gfx1250__)
  extern __shared__ char smem[];
  const int tid   = threadIdx.x;
  const int bx    = blockIdx.x;
  const int ob    = bx & 7;          // O-block (8 x 128)
  const int kv    = bx >> 3;
  const int kk    = kv >> 4;         // k index (0..3)
  const int v     = kv & 15;         // selected-group slot (0..15)
  const int obase = ob * OBLK;
  const int g     = gidx[v];

  const float* xbase = x + ((size_t)kk * VSEL + v) * IN_SIZE;              // b-stride = K*V*I
  const float* wbase = w + ((size_t)(kk * NGROUPS + g) * OUT_SIZE + obase) * IN_SIZE;

  const int lane = tid & 31;
  const int wv   = tid >> 5;         // wave = N-tile (0..7)
  const int lh   = lane >> 4;        // lane half
  const int lm   = lane & 15;

  v8f acc[16];
  const v8f zero = {0.f,0.f,0.f,0.f,0.f,0.f,0.f,0.f};
#pragma unroll
  for (int m = 0; m < 16; ++m) acc[m] = zero;

#if HAS_TDM
  const unsigned lds0 = (unsigned)(unsigned long long)(void*)smem;
  auto issue = [&](int c, int b) {
    const unsigned off = lds0 + (unsigned)(b * BUFBYTES);
    tdm_load_2d(off,          xbase + (size_t)c * KC, KC, XROWS,
                (unsigned long long)KDIM * VSEL * IN_SIZE);
    tdm_load_2d(off + XBYTES, wbase + (size_t)c * KC, KC, WROWS,
                (unsigned long long)IN_SIZE);
  };
  if (tid < 32) issue(0, 0);         // prologue (wave 0 drives the TDM)
#endif

  for (int c = 0; c < NCHUNK; ++c) {
    char* buf = smem + (c & 1) * BUFBYTES;

#if HAS_TDM
    if (tid < 32) {
      if (c + 1 < NCHUNK) {
        issue(c + 1, (c + 1) & 1);                 // overlap next chunk's DMA
        __builtin_amdgcn_s_wait_tensorcnt(2);      // chunk c's 2 tensor ops done
      } else {
        __builtin_amdgcn_s_wait_tensorcnt(0);
      }
    }
#else
    // Fallback staging: global -> LDS with the same padded layout.
    {
      const float* xs = xbase + (size_t)tid * (KDIM * VSEL * IN_SIZE) + (size_t)c * KC;
      float4v* dx = (float4v*)(buf + tid * ROWB);
#pragma unroll
      for (int j = 0; j < 8; ++j) dx[j] = ((const float4v*)xs)[j];
      if (tid < WROWS) {
        const float* ws = wbase + (size_t)tid * IN_SIZE + (size_t)c * KC;
        float4v* dw = (float4v*)(buf + XBYTES + tid * ROWB);
#pragma unroll
        for (int j = 0; j < 8; ++j) dw[j] = ((const float4v*)ws)[j];
      }
    }
#endif
    __syncthreads();

    // fp32 -> (bf16 hi | bf16 lo) split, in place (bf16x3 decomposition)
    cvt_row(buf + tid * ROWB);                       // X: 256 rows / 256 threads
    if (tid < WROWS) cvt_row(buf + XBYTES + tid * ROWB); // W: 128 rows
    __syncthreads();

    // ---- WMMA compute: D = Xh*Wh + Xh*Wl + Xl*Wh (f32 accumulate) ----
    // B (32x16 bf16): lane = {Krun = lh*16.., N = lm}; contiguous 32B per lane.
    const char* wrow = buf + XBYTES + (wv * 16 + lm) * ROWB;
    const int bko = lh * 32;
    v16bf Bh = load_frag(wrow,      bko, bko + 16);
    v16bf Bl = load_frag(wrow + 64, bko, bko + 16);

#pragma unroll
    for (int m = 0; m < 16; ++m) {
      // A (16x32 bf16): lane holds row M=lm, K-run [lh*16..+15]
      const char* xrow = buf + (m * 16 + lm) * ROWB;
      const int ao = lh * 32;
      v16bf Ah = load_frag(xrow,      ao, ao + 16);
      v16bf Al = load_frag(xrow + 64, ao, ao + 16);
      acc[m] = __builtin_amdgcn_wmma_f32_16x16x32_bf16(false, Ah, false, Bh,
                                                       (short)0, acc[m], false, false);
      acc[m] = __builtin_amdgcn_wmma_f32_16x16x32_bf16(false, Ah, false, Bl,
                                                       (short)0, acc[m], false, false);
      acc[m] = __builtin_amdgcn_wmma_f32_16x16x32_bf16(false, Al, false, Bh,
                                                       (short)0, acc[m], false, false);
    }
    __syncthreads();   // everyone done with buf before TDM reuses it
  }

  // ---- epilogue: bias + store ----
  // C/D layout: lanes 0-15: vgpr r -> (M=r, N=lane); lanes 16-31: (M=8+r, N=lane-16)
  const int ocol = obase + wv * 16 + lm;
  const float bv = bias[(size_t)g * OUT_SIZE + ocol];
#pragma unroll
  for (int m = 0; m < 16; ++m) {
#pragma unroll
    for (int r = 0; r < 8; ++r) {
      const int brow = m * 16 + lh * 8 + r;
      out[(((size_t)brow * KDIM + kk) * VSEL + v) * OUT_SIZE + ocol] = acc[m][r] + bv;
    }
  }
#else
  (void)x; (void)w; (void)bias; (void)gidx; (void)out;
#endif
}

extern "C" void kernel_launch(void* const* d_in, const int* in_sizes, int n_in,
                              void* d_out, int out_size, void* d_ws, size_t ws_size,
                              hipStream_t stream) {
  (void)in_sizes; (void)n_in; (void)out_size; (void)d_ws; (void)ws_size;
  const float* x    = (const float*)d_in[0];
  const float* w    = (const float*)d_in[1];
  const float* bias = (const float*)d_in[2];
  const int*   gi   = (const int*)d_in[3];
  float* out = (float*)d_out;

  dim3 grid((OUT_SIZE / OBLK) * KDIM * VSEL);   // 8 * 64 = 512 workgroups
  dim3 block(256);                              // 8 waves (wave32)
  kgrouped_linear_wmma<<<grid, block, SMEM_TOTAL, stream>>>(x, w, bias, gi, out);
}
